// LocallyConnectedNetwork_58171037057774
// MI455X (gfx1250) — matrix operand order repaired and checked by reading, not compiled, not run on hardware
//
#include <hip/hip_runtime.h>
#include <hip/hip_bf16.h>
#include <stdint.h>

// ---------------------------------------------------------------------------
// Types for CDNA5 WMMA (wave32): v16bf = 16 bf16 (8 VGPRs), v8f = 8 f32.
// ---------------------------------------------------------------------------
typedef __attribute__((ext_vector_type(16))) __bf16 v16bf;
typedef __attribute__((ext_vector_type(8)))  float  v8f;

// Problem dims
#define B_    128
#define C_    3
#define H_    64
#define W_    64
#define OC_   64
#define OH_   32
#define OW_   32
#define NLOC  1024        // OH_*OW_
#define LC_K  75          // C*K*K
#define LC_KP 96          // padded to multiple of 32
#define FCIN  65536       // OC*OH*OW
#define HID   1536
#define NCLS  10
#define EPS_  1e-5f

// ---------------------------------------------------------------------------
// CDNA5 async global->LDS copy (ASYNCcnt path, ISA 10. / 15.18.3 op 98).
// LDS destination address = low 32 bits of the generic (flat) address of the
// __shared__ element (flat LDS aperture keeps the byte offset in addr[31:0]).
// ---------------------------------------------------------------------------
__device__ __forceinline__ void async_copy16(void* lds_dst, const void* gsrc) {
  uint32_t lds_off = (uint32_t)(uintptr_t)lds_dst;
  asm volatile("global_load_async_to_lds_b128 %0, %1, off"
               :
               : "v"(lds_off), "v"(gsrc)
               : "memory");
}

__device__ __forceinline__ void wait_async0() {
#if defined(__has_builtin) && __has_builtin(__builtin_amdgcn_s_wait_asynccnt)
  __builtin_amdgcn_s_wait_asynccnt(0);
#else
  asm volatile("s_wait_asynccnt 0" ::: "memory");
#endif
}

// ---------------------------------------------------------------------------
// WMMA fragment loaders (layouts from CDNA5 ISA 7.12.2)
// A (16x32 bf16): lane l -> row m = m0 + (l&15); per-lane K chunks:
//   elems 0..7  = K = k0 + 8*hi + (0..7)
//   elems 8..15 = K = k0 + 16 + 8*hi + (0..7)          (hi = l>>4)
// ---------------------------------------------------------------------------
__device__ __forceinline__ v16bf load_frag_a(const __bf16* Am, int ldk,
                                             int m0, int k0, int lane) {
  int m  = m0 + (lane & 15);
  int hi = (lane >> 4) * 8;
  const __bf16* row = Am + m * ldk + k0;
  v16bf a;
#pragma unroll
  for (int j = 0; j < 8; ++j) {
    a[j]     = row[hi + j];
    a[j + 8] = row[16 + hi + j];
  }
  return a;
}

// B (32x16 bf16), stored in LDS as [n][k] (n-major, stride ldk):
// lane l -> col n = n0 + (l&15); per-lane K chunks (same as A):
// elems 0..7 = K = k0 + 8*hi + (0..7), elems 8..15 = K = k0 + 16 + 8*hi + (0..7).
__device__ __forceinline__ v16bf load_frag_b(const __bf16* Bm, int ldk,
                                             int n0, int k0, int lane) {
  int n = n0 + (lane & 15);
  int hi = (lane >> 4) * 8;
  const __bf16* row = Bm + n * ldk + k0;
  v16bf b;
#pragma unroll
  for (int j = 0; j < 8; ++j) {
    b[j]     = row[hi + j];
    b[j + 8] = row[16 + hi + j];
  }
  return b;
}

// ---------------------------------------------------------------------------
// Kernel 0: zero the split-K accumulator h [B, HID]
// ---------------------------------------------------------------------------
__global__ __launch_bounds__(256) void k_zero(float* p, int n) {
  int i = blockIdx.x * 256 + threadIdx.x;
  if (i < n) p[i] = 0.f;
}

// ---------------------------------------------------------------------------
// Kernel 1: LocallyConnected2d via per-location WMMA GEMM.
// One block per output location. A = patches [128 x 75] (gathered from x),
// B = lc_w[:, :, oh, ow] [64 x 75]; K padded to 96. 8 waves, each wave owns
// one 16-row M tile and all 4 N tiles -> 96 v_wmma per block.
// ---------------------------------------------------------------------------
__global__ __launch_bounds__(256) void k_lc(const float* __restrict__ x,
                                            const float* __restrict__ lcw,
                                            float* __restrict__ out_pre) {
  __shared__ __align__(32) __bf16 Al[B_ * LC_KP];   // 24576 B
  __shared__ __align__(32) __bf16 Bl[OC_ * LC_KP];  // 12288 B

  int loc = blockIdx.x;              // 0..1023
  int oh = loc >> 5, ow = loc & 31;
  int tid = threadIdx.x;

  // Gather patch panel A (implicit im2col), bf16, zero-padded K
  for (int i = tid; i < B_ * LC_KP; i += 256) {
    int b = i / LC_KP, k = i % LC_KP;
    float v = 0.f;
    if (k < LC_K) {
      int c = k / 25, r = k % 25;
      int kh = r / 5, kw = r % 5;
      int ih = oh * 2 - 2 + kh;
      int iw = ow * 2 - 2 + kw;
      if ((unsigned)ih < (unsigned)H_ && (unsigned)iw < (unsigned)W_)
        v = x[((b * C_ + c) * H_ + ih) * W_ + iw];
    }
    Al[i] = (__bf16)v;
  }
  // Per-location weight panel B: lc_w[o][k][oh][ow], stride over k is 1024
  for (int i = tid; i < OC_ * LC_KP; i += 256) {
    int o = i / LC_KP, k = i % LC_KP;
    float v = (k < LC_K) ? lcw[(size_t)(o * LC_K + k) * NLOC + loc] : 0.f;
    Bl[i] = (__bf16)v;
  }
  __syncthreads();

  int wave = tid >> 5, lane = tid & 31;
  int m0 = wave * 16;
  v8f acc[4] = {};
#pragma unroll
  for (int k0 = 0; k0 < LC_KP; k0 += 32) {
    v16bf a = load_frag_a(Al, LC_KP, m0, k0, lane);
#pragma unroll
    for (int nt = 0; nt < 4; ++nt) {
      v16bf b = load_frag_b(Bl, LC_KP, nt * 16, k0, lane);
      acc[nt] = __builtin_amdgcn_wmma_f32_16x16x32_bf16(
          false, a, false, b, (short)0, acc[nt], false, false);
    }
  }

  // Store: out_pre[b][o][oh][ow];  m = batch, n = channel
  int hi = lane >> 4, nn = lane & 15;
#pragma unroll
  for (int nt = 0; nt < 4; ++nt) {
    int o = nt * 16 + nn;
#pragma unroll
    for (int r = 0; r < 8; ++r) {
      int b = m0 + r + 8 * hi;
      out_pre[(size_t)((b * OC_ + o) << 10) + loc] = acc[nt][r];
    }
  }
}

// ---------------------------------------------------------------------------
// Kernel 2: BN2d batch statistics -> per-channel scale/shift
// ---------------------------------------------------------------------------
__global__ __launch_bounds__(256) void k_bn1_stats(const float* __restrict__ pre,
                                                   const float* __restrict__ g,
                                                   const float* __restrict__ bta,
                                                   float* __restrict__ sc,
                                                   float* __restrict__ sh) {
  __shared__ float rs[256], rss[256];
  int c = blockIdx.x, tid = threadIdx.x;
  float s = 0.f, ss = 0.f;
  for (int i = tid; i < B_ * NLOC; i += 256) {
    int b = i >> 10, loc = i & 1023;
    float v = pre[(size_t)((b * OC_ + c) << 10) + loc];
    s += v; ss += v * v;
  }
  rs[tid] = s; rss[tid] = ss;
  __syncthreads();
  for (int st = 128; st > 0; st >>= 1) {
    if (tid < st) { rs[tid] += rs[tid + st]; rss[tid] += rss[tid + st]; }
    __syncthreads();
  }
  if (tid == 0) {
    const float inv = 1.f / (float)(B_ * NLOC);
    float mean = rs[0] * inv;
    float var  = rss[0] * inv - mean * mean;
    float scale = g[c] * rsqrtf(var + EPS_);
    sc[c] = scale;
    sh[c] = bta[c] - mean * scale;
  }
}

// ---------------------------------------------------------------------------
// Kernel 3: normalize + ReLU + fp32->bf16, producing act1 [B, FCIN]
// (flat layout of out_pre is already [b][o*1024+loc] == reshape(B,-1))
// ---------------------------------------------------------------------------
__global__ __launch_bounds__(256) void k_bn1_apply(const float* __restrict__ pre,
                                                   const float* __restrict__ sc,
                                                   const float* __restrict__ sh,
                                                   __bf16* __restrict__ act) {
  int i = (blockIdx.x * 256 + threadIdx.x) * 4;
#pragma unroll
  for (int j = 0; j < 4; ++j) {
    int e = i + j;
    int o = (e >> 10) & (OC_ - 1);
    float v = pre[e] * sc[o] + sh[o];
    act[e] = (__bf16)(v > 0.f ? v : 0.f);
  }
}

// ---------------------------------------------------------------------------
// Kernel 4: FC1 GEMM  h[128,1536] += act1[128,65536] @ fc1_w[1536,65536]^T
// Split-K: grid = (24 N-tiles of 64, 32 K-partitions of 2048).
// Double-buffered K-panels of 128:
//   - act panel (32 KB)  : async DMA global->LDS (global_load_async_to_lds_b128,
//                          ASYNCcnt), issued for panel p+1 while computing p
//   - weight panel (16KB): fp32 loaded to VGPRs for panel p+1 during compute,
//                          converted to bf16 and stored to LDS after the WMMAs
// fc1_w (402 MB) is read exactly once -> HBM-roofline bound (~17 us floor).
// Each wave: 16 panels x 4 ksteps x 4 ntiles = 256 v_wmma.
// ---------------------------------------------------------------------------
#define KPAN 128
#define KCHUNK 2048
#define NPAN (KCHUNK / KPAN)
__global__ __launch_bounds__(256) void k_fc1(const __bf16* __restrict__ act,
                                             const float* __restrict__ w,
                                             float* __restrict__ h) {
  __shared__ __align__(16) __bf16 Al[2][B_ * KPAN];   // 2 x 32 KB
  __shared__ __align__(16) __bf16 Wl[2][64 * KPAN];   // 2 x 16 KB

  int tid  = threadIdx.x;
  int n0   = blockIdx.x * 64;          // 0..1535 step 64
  int korg = blockIdx.y * KCHUNK;      // 0..65535 step 2048
  int wave = tid >> 5, lane = tid & 31;
  int m0   = wave * 16;

  // Per-thread copy assignments
  int arow  = tid >> 1, ahalf = (tid & 1) * 64;   // act: 8 x 16B async chunks
  int wrow  = tid >> 2, wq    = (tid & 3) * 32;   // weights: 32 f32 (8 float4)
  const __bf16* abase = act + (size_t)arow * FCIN + ahalf;
  const float*  wbase = w + (size_t)(n0 + wrow) * FCIN + wq;

  // ---- prologue: stage panel 0
  {
#pragma unroll
    for (int j = 0; j < 8; ++j)
      async_copy16(&Al[0][arow * KPAN + ahalf + j * 8], abase + korg + j * 8);
    const float4* ws4 = (const float4*)(wbase + korg);
    __bf16* dstw = &Wl[0][wrow * KPAN + wq];
#pragma unroll
    for (int j = 0; j < 8; ++j) {
      float4 f = ws4[j];
      dstw[4 * j + 0] = (__bf16)f.x;
      dstw[4 * j + 1] = (__bf16)f.y;
      dstw[4 * j + 2] = (__bf16)f.z;
      dstw[4 * j + 3] = (__bf16)f.w;
    }
    wait_async0();
  }
  __syncthreads();

  v8f acc[4] = {};
  for (int p = 0; p < NPAN; ++p) {
    int cur = p & 1, nxt = cur ^ 1;
    bool more = (p + 1) < NPAN;

    // Issue next panel's traffic before touching the WMMAs:
    float4 f[8];
    if (more) {
      int kb = korg + (p + 1) * KPAN;
#pragma unroll
      for (int j = 0; j < 8; ++j)
        async_copy16(&Al[nxt][arow * KPAN + ahalf + j * 8], abase + kb + j * 8);
      const float4* ws4 = (const float4*)(wbase + kb);
#pragma unroll
      for (int j = 0; j < 8; ++j) f[j] = ws4[j];
    }

    // Compute on current panel while the async DMA + weight loads fly
#pragma unroll
    for (int ks = 0; ks < KPAN; ks += 32) {
      v16bf a = load_frag_a(&Al[cur][0], KPAN, m0, ks, lane);
#pragma unroll
      for (int nt = 0; nt < 4; ++nt) {
        v16bf b = load_frag_b(&Wl[cur][0], KPAN, nt * 16, ks, lane);
        acc[nt] = __builtin_amdgcn_wmma_f32_16x16x32_bf16(
            false, a, false, b, (short)0, acc[nt], false, false);
      }
    }

    // Convert + park next weight panel, drain our async loads, publish
    if (more) {
      __bf16* dstw = &Wl[nxt][wrow * KPAN + wq];
#pragma unroll
      for (int j = 0; j < 8; ++j) {
        float4 ff = f[j];
        dstw[4 * j + 0] = (__bf16)ff.x;
        dstw[4 * j + 1] = (__bf16)ff.y;
        dstw[4 * j + 2] = (__bf16)ff.z;
        dstw[4 * j + 3] = (__bf16)ff.w;
      }
      wait_async0();
    }
    __syncthreads();
  }

  // Split-K accumulate
  int hi = lane >> 4, nn = lane & 15;
#pragma unroll
  for (int nt = 0; nt < 4; ++nt) {
    int n = n0 + nt * 16 + nn;
#pragma unroll
    for (int r = 0; r < 8; ++r) {
      int b = m0 + r + 8 * hi;
      atomicAdd(&h[(size_t)b * HID + n], acc[nt][r]);
    }
  }
}

// ---------------------------------------------------------------------------
// Kernel 5: BN1d stats over batch -> per-feature scale/shift
// ---------------------------------------------------------------------------
__global__ __launch_bounds__(256) void k_bn2_stats(const float* __restrict__ h,
                                                   const float* __restrict__ g,
                                                   const float* __restrict__ bta,
                                                   float* __restrict__ sc,
                                                   float* __restrict__ sh) {
  int col = blockIdx.x * 256 + threadIdx.x;
  if (col >= HID) return;
  float s = 0.f, ss = 0.f;
  for (int b = 0; b < B_; ++b) {
    float v = h[(size_t)b * HID + col];
    s += v; ss += v * v;
  }
  const float inv = 1.f / (float)B_;
  float mean = s * inv;
  float var  = ss * inv - mean * mean;
  float scale = g[col] * rsqrtf(var + EPS_);
  sc[col] = scale;
  sh[col] = bta[col] - mean * scale;
}

// ---------------------------------------------------------------------------
// Kernel 6: head — normalize+ReLU a batch row in LDS, then 10 dot products
// ---------------------------------------------------------------------------
__global__ __launch_bounds__(256) void k_head(const float* __restrict__ h,
                                              const float* __restrict__ sc,
                                              const float* __restrict__ sh,
                                              const float* __restrict__ fc2w,
                                              float* __restrict__ out) {
  __shared__ float hn[HID];
  __shared__ float red[256];
  int b = blockIdx.x, tid = threadIdx.x;
  for (int i = tid; i < HID; i += 256) {
    float v = h[(size_t)b * HID + i] * sc[i] + sh[i];
    hn[i] = v > 0.f ? v : 0.f;
  }
  __syncthreads();
  for (int cls = 0; cls < NCLS; ++cls) {
    float p = 0.f;
    for (int k = tid; k < HID; k += 256) p += hn[k] * fc2w[cls * HID + k];
    red[tid] = p;
    __syncthreads();
    for (int st = 128; st > 0; st >>= 1) {
      if (tid < st) red[tid] += red[tid + st];
      __syncthreads();
    }
    if (tid == 0) out[b * NCLS + cls] = red[0];
    __syncthreads();
  }
}

// ---------------------------------------------------------------------------
// Workspace layout (bytes)
// ---------------------------------------------------------------------------
#define OFF_OUTPRE  ((size_t)0)                          // 8388608 f32
#define OFF_ACT1    (OFF_OUTPRE + (size_t)8388608 * 4)   // 8388608 bf16
#define OFF_H       (OFF_ACT1   + (size_t)8388608 * 2)   // 196608 f32
#define OFF_BN1SC   (OFF_H      + (size_t)196608 * 4)    // 64 f32
#define OFF_BN1SH   (OFF_BN1SC  + 256)
#define OFF_BN2SC   (OFF_BN1SH  + 256)                   // 1536 f32
#define OFF_BN2SH   (OFF_BN2SC  + 6144)

extern "C" void kernel_launch(void* const* d_in, const int* in_sizes, int n_in,
                              void* d_out, int out_size, void* d_ws, size_t ws_size,
                              hipStream_t stream) {
  const float* x     = (const float*)d_in[0];
  const float* lcw   = (const float*)d_in[1];
  const float* bn1g  = (const float*)d_in[2];
  const float* bn1b  = (const float*)d_in[3];
  const float* fc1w  = (const float*)d_in[4];
  const float* bn2g  = (const float*)d_in[5];
  const float* bn2b  = (const float*)d_in[6];
  const float* fc2w  = (const float*)d_in[7];
  float* out = (float*)d_out;

  char* ws = (char*)d_ws;
  float*  out_pre = (float*)(ws + OFF_OUTPRE);
  __bf16* act1    = (__bf16*)(ws + OFF_ACT1);
  float*  h       = (float*)(ws + OFF_H);
  float*  bn1sc   = (float*)(ws + OFF_BN1SC);
  float*  bn1sh   = (float*)(ws + OFF_BN1SH);
  float*  bn2sc   = (float*)(ws + OFF_BN2SC);
  float*  bn2sh   = (float*)(ws + OFF_BN2SH);

  // 0) zero split-K accumulator (atomics require it every call)
  k_zero<<<(B_ * HID + 255) / 256, 256, 0, stream>>>(h, B_ * HID);

  // 1) LocallyConnected2d (WMMA, one block per location)
  k_lc<<<NLOC, 256, 0, stream>>>(x, lcw, out_pre);

  // 2) BN2d stats -> scale/shift
  k_bn1_stats<<<OC_, 256, 0, stream>>>(out_pre, bn1g, bn1b, bn1sc, bn1sh);

  // 3) normalize + ReLU + bf16 activations
  k_bn1_apply<<<(B_ * FCIN) / (256 * 4), 256, 0, stream>>>(out_pre, bn1sc, bn1sh, act1);

  // 4) FC1 (WMMA, split-K, async double-buffered)
  dim3 g4(HID / 64, FCIN / KCHUNK);
  k_fc1<<<g4, 256, 0, stream>>>(act1, fc1w, h);

  // 5) BN1d stats
  k_bn2_stats<<<HID / 256, 256, 0, stream>>>(h, bn2g, bn2b, bn2sc, bn2sh);

  // 6) BN + ReLU + FC2
  k_head<<<B_, 256, 0, stream>>>(h, bn2sc, bn2sh, fc2w, out);
}
